// CausalSelfAttention_88227218194532
// MI455X (gfx1250) — matrix operand order repaired and checked by reading, not compiled, not run on hardware
//
#include <hip/hip_runtime.h>

// ---------------------------------------------------------------------------
// CDNA5 (gfx1250) causal self-attention: bf16 WMMA, fp32 softmax,
// GLOBAL_LOAD_ASYNC_TO_LDS staging for bf16 tiles.
//   B=4, T=2048, C=1024, NH=16, HD=64
// Pipeline: qkv_gemm (fp32->bf16, WMMA) -> flash attention (WMMA) -> proj_gemm
// Workspace: Q,K,V ([B,NH,T,HD] bf16) + Y ([B,T,C] bf16) = 64 MB.
// ---------------------------------------------------------------------------

typedef __attribute__((ext_vector_type(16))) __bf16        v16bf;
typedef __attribute__((ext_vector_type(2)))  __bf16        v2bf;
typedef __attribute__((ext_vector_type(8)))  float         v8f;
typedef __attribute__((ext_vector_type(8)))  unsigned int  v8u;

#define Bq  4
#define Tq  2048
#define Cq  1024
#define NHq 16
#define HDq 64

__device__ __forceinline__ unsigned short f2bf(float f) {
  return __builtin_bit_cast(unsigned short, (__bf16)f);
}
__device__ __forceinline__ unsigned int pack2bf(float lo, float hi) {
  v2bf t;
  t[0] = (__bf16)lo;
  t[1] = (__bf16)hi;
  return __builtin_bit_cast(unsigned int, t);
}

// Async DMA: global -> LDS, 16 bytes per lane, tracked by ASYNCcnt.
__device__ __forceinline__ void async_ld16(const unsigned short* g,
                                           unsigned short* l) {
  unsigned int       la = (unsigned int)(unsigned long long)l;   // LDS addr
  unsigned long long ga = (unsigned long long)g;                 // global addr
  asm volatile("global_load_async_to_lds_b128 %0, %1, off"
               :: "v"(la), "v"(ga) : "memory");
}
__device__ __forceinline__ void wait_async0() {
  asm volatile("s_wait_asynccnt 0x0" ::: "memory");
}

// A-fragment (16x32 bf16, MxK), ISA 7.12.2:
//  lanes 0-15 : row M=lane,    K = {0..7, 16..23}   (pairs packed in u32)
//  lanes 16-31: row M=lane-16, K = {8..15, 24..31}
__device__ __forceinline__ v16bf frag_a(const unsigned short* p, int stride,
                                        int m0, int k0) {
  const int lane = threadIdx.x & 31;
  const int half = (lane >> 4) & 1;
  const int m    = m0 + (lane & 15);
  const int kb   = k0 + half * 8;                 // even
  const unsigned int* q = (const unsigned int*)(p + m * stride);
  v8u r;
  r[0] = q[(kb >> 1) + 0];  r[1] = q[(kb >> 1) + 1];
  r[2] = q[(kb >> 1) + 2];  r[3] = q[(kb >> 1) + 3];
  r[4] = q[(kb >> 1) + 8];  r[5] = q[(kb >> 1) + 9];
  r[6] = q[(kb >> 1) + 10]; r[7] = q[(kb >> 1) + 11];
  return __builtin_bit_cast(v16bf, r);
}

// B-fragment (32x16 bf16, KxN) read from LDS stored as Bt[n][k] (k contiguous):
//  lanes 0-15 : col N=lane, K={0..7,16..23} ; lanes 16-31: col N=lane-16, K={8..15,24..31}
__device__ __forceinline__ v16bf frag_b(const unsigned short* p, int stride,
                                        int n0, int k0) {
  const int lane = threadIdx.x & 31;
  const int n    = n0 + (lane & 15);
  const int kb   = k0 + ((lane >> 4) & 1) * 8;    // even
  const unsigned int* q = (const unsigned int*)(p + n * stride);
  v8u r;
#pragma unroll
  for (int j = 0; j < 8; ++j) r[j] = q[(kb >> 1) + (j & 3) + (j >> 2) * 8];
  return __builtin_bit_cast(v16bf, r);
}

__device__ __forceinline__ v8f wmma_bf16(v16bf a, v16bf b, v8f c) {
  return __builtin_amdgcn_wmma_f32_16x16x32_bf16(
      false, a, false, b, (short)0, c, false, false);
}

// ---------------------------------------------------------------------------
// Kernel 1: qkv = x @ w_qkv  (fp32 in, bf16 WMMA, bf16 out scattered to Q/K/V)
// Block tile 128x64, K-step 32. 128 threads = 4 waves; wave owns 32 rows.
// ---------------------------------------------------------------------------
__global__ __launch_bounds__(128)
void qkv_gemm_kernel(const float* __restrict__ X, const float* __restrict__ W,
                     unsigned short* __restrict__ Qo,
                     unsigned short* __restrict__ Ko,
                     unsigned short* __restrict__ Vo) {
  __shared__ __align__(16) unsigned short As[128 * 40];  // [m][k], stride 40
  __shared__ __align__(16) unsigned short Bs[64 * 40];   // [n][k]
  const int tid = threadIdx.x, lane = tid & 31, wave = tid >> 5;
  const int half = (lane >> 4) & 1;
  const int m0 = blockIdx.y * 128;
  const int n0 = blockIdx.x * 64;

  v8f acc[2][4] = {};
  unsigned int* As32 = (unsigned int*)As;
  for (int k0 = 0; k0 < Cq; k0 += 32) {
    __syncthreads();
#pragma unroll
    for (int i = 0; i < 16; ++i) {
      int idx = tid + i * 128;                 // 0..2047
      int m = idx >> 4, kp = idx & 15;         // A: 128 x 16 float-pairs
      const float* src = X + (size_t)(m0 + m) * Cq + k0 + kp * 2;
      As32[m * 20 + kp] = pack2bf(src[0], src[1]);
      int kk = idx >> 6, n = idx & 63;         // B: 32 x 64, store transposed
      Bs[n * 40 + kk] = f2bf(W[(size_t)(k0 + kk) * (3 * Cq) + n0 + n]);
    }
    __syncthreads();
    v16bf a0 = frag_a(As, 40, wave * 32, 0);
    v16bf a1 = frag_a(As, 40, wave * 32 + 16, 0);
#pragma unroll
    for (int nt = 0; nt < 4; ++nt) {
      v16bf b = frag_b(Bs, 40, nt * 16, 0);
      acc[0][nt] = wmma_bf16(a0, b, acc[0][nt]);
      acc[1][nt] = wmma_bf16(a1, b, acc[1][nt]);
    }
  }
  // scatter to [B, NH, T, HD] bf16
#pragma unroll
  for (int rt = 0; rt < 2; ++rt) {
#pragma unroll
    for (int nt = 0; nt < 4; ++nt) {
      int gn = n0 + nt * 16 + (lane & 15);
      int part = gn >> 10, cc = gn & 1023, h = cc >> 6, hd = cc & 63;
      unsigned short* base = (part == 0) ? Qo : (part == 1 ? Ko : Vo);
#pragma unroll
      for (int e = 0; e < 8; ++e) {
        int gm = m0 + wave * 32 + rt * 16 + e + 8 * half;
        int b = gm >> 11, t = gm & 2047;
        base[((size_t)(b * NHq + h) * Tq + t) * HDq + hd] = f2bf(acc[rt][nt][e]);
      }
    }
  }
}

// ---------------------------------------------------------------------------
// Kernel 2: flash attention. Grid (T/64, NH, B), 128 thr = 4 waves.
// K tile staged via GLOBAL_LOAD_ASYNC_TO_LDS; V staged transposed via VGPRs.
// ---------------------------------------------------------------------------
__global__ __launch_bounds__(128)
void attn_kernel(const unsigned short* __restrict__ Qb,
                 const unsigned short* __restrict__ Kb,
                 const unsigned short* __restrict__ Vb,
                 unsigned short* __restrict__ Yb) {
  __shared__ __align__(16) unsigned short Ks[64 * 72];      // [key][hd]
  __shared__ __align__(16) unsigned short Vt[64 * 72];      // [hd][key]
  __shared__ __align__(16) unsigned short Ps[4 * 16 * 72];  // per-wave P tile
  const int tid = threadIdx.x, lane = tid & 31, wave = tid >> 5;
  const int half = (lane >> 4) & 1;
  const int qt = blockIdx.x, h = blockIdx.y, b = blockIdx.z;
  const unsigned short* Qh = Qb + (size_t)(b * NHq + h) * Tq * HDq;
  const unsigned short* Kh = Kb + (size_t)(b * NHq + h) * Tq * HDq;
  const unsigned short* Vh = Vb + (size_t)(b * NHq + h) * Tq * HDq;
  const int qr0 = qt * 64 + wave * 16;

  v16bf qf0 = frag_a(Qh, HDq, qr0, 0);
  v16bf qf1 = frag_a(Qh, HDq, qr0, 32);

  v8f oacc[4] = {};
  float rmax[8], rsum[8];
#pragma unroll
  for (int e = 0; e < 8; ++e) { rmax[e] = -1e30f; rsum[e] = 0.f; }

  unsigned short* Pw = Ps + wave * 16 * 72;

  for (int j = 0; j <= qt; ++j) {   // causal: only tiles at/below the diagonal
    const int kb0 = j * 64;
    __syncthreads();
    // K tile: 64 rows x 128B, async DMA to LDS (stride 144B, 16B segments)
#pragma unroll
    for (int i = 0; i < 4; ++i) {
      int idx = tid + i * 128;                  // 0..511 16B segments
      int key = idx >> 3, seg = idx & 7;
      async_ld16(Kh + (size_t)(kb0 + key) * HDq + seg * 8,
                 &Ks[key * 72 + seg * 8]);
    }
    // V tile transposed via VGPRs
#pragma unroll
    for (int i = 0; i < 32; ++i) {
      int idx = tid + i * 128;                  // 0..4095
      int key = idx >> 6, hd = idx & 63;
      Vt[hd * 72 + key] = Vh[(size_t)(kb0 + key) * HDq + hd];
    }
    wait_async0();
    __syncthreads();

    // S = Q @ K^T   (4 key sub-tiles x 2 K-steps over HD)
    v8f sacc[4];
#pragma unroll
    for (int nt = 0; nt < 4; ++nt) {
      v8f c = {};
      c = wmma_bf16(qf0, frag_b(Ks, 72, nt * 16, 0), c);
      c = wmma_bf16(qf1, frag_b(Ks, 72, nt * 16, 32), c);
      sacc[nt] = c;
    }
    const bool diag = (j == qt);
#pragma unroll
    for (int nt = 0; nt < 4; ++nt) {
      int key = kb0 + nt * 16 + (lane & 15);
#pragma unroll
      for (int e = 0; e < 8; ++e) {
        float s = sacc[nt][e] * 0.125f;         // 1/sqrt(64)
        int qrow = qr0 + e + 8 * half;
        if (diag && key > qrow) s = -1e30f;
        sacc[nt][e] = s;
      }
    }
    // online softmax: row M = e + 8*half spans the 16 lanes of each half
#pragma unroll
    for (int e = 0; e < 8; ++e) {
      float m = fmaxf(fmaxf(sacc[0][e], sacc[1][e]),
                      fmaxf(sacc[2][e], sacc[3][e]));
      m = fmaxf(m, __shfl_xor(m, 1));
      m = fmaxf(m, __shfl_xor(m, 2));
      m = fmaxf(m, __shfl_xor(m, 4));
      m = fmaxf(m, __shfl_xor(m, 8));
      float mnew = fmaxf(rmax[e], m);
      float corr = __expf(rmax[e] - mnew);
      rmax[e] = mnew;
      float ps = 0.f;
#pragma unroll
      for (int nt = 0; nt < 4; ++nt) {
        float p = __expf(sacc[nt][e] - mnew);
        sacc[nt][e] = p;
        ps += p;
      }
      ps += __shfl_xor(ps, 1); ps += __shfl_xor(ps, 2);
      ps += __shfl_xor(ps, 4); ps += __shfl_xor(ps, 8);
      rsum[e] = rsum[e] * corr + ps;
#pragma unroll
      for (int nt = 0; nt < 4; ++nt) oacc[nt][e] *= corr;
    }
    // reshape P through LDS into A-fragment layout
#pragma unroll
    for (int nt = 0; nt < 4; ++nt) {
      int col = nt * 16 + (lane & 15);
#pragma unroll
      for (int e = 0; e < 8; ++e)
        Pw[(e + 8 * half) * 72 + col] = f2bf(sacc[nt][e]);
    }
    __syncthreads();
    v16bf p0 = frag_a(Pw, 72, 0, 0);
    v16bf p1 = frag_a(Pw, 72, 0, 32);
#pragma unroll
    for (int nt = 0; nt < 4; ++nt) {
      oacc[nt] = wmma_bf16(p0, frag_b(Vt, 72, nt * 16, 0), oacc[nt]);
      oacc[nt] = wmma_bf16(p1, frag_b(Vt, 72, nt * 16, 32), oacc[nt]);
    }
  }
  // O = acc / rowsum -> Y [B,T,C] bf16
#pragma unroll
  for (int nt = 0; nt < 4; ++nt) {
    int col = h * HDq + nt * 16 + (lane & 15);
#pragma unroll
    for (int e = 0; e < 8; ++e) {
      int trow = qr0 + e + 8 * half;
      Yb[((size_t)b * Tq + trow) * Cq + col] = f2bf(oacc[nt][e] / rsum[e]);
    }
  }
}

// ---------------------------------------------------------------------------
// Kernel 3: out = y @ w_proj. A tile (bf16) staged via async DMA; 128x64 tile.
// ---------------------------------------------------------------------------
__global__ __launch_bounds__(128)
void proj_gemm_kernel(const unsigned short* __restrict__ Ybf,
                      const float* __restrict__ W, float* __restrict__ Out) {
  __shared__ __align__(16) unsigned short As[128 * 40];
  __shared__ __align__(16) unsigned short Bs[64 * 40];
  const int tid = threadIdx.x, lane = tid & 31, wave = tid >> 5;
  const int half = (lane >> 4) & 1;
  const int m0 = blockIdx.y * 128;
  const int n0 = blockIdx.x * 64;

  v8f acc[2][4] = {};
  for (int k0 = 0; k0 < Cq; k0 += 32) {
    __syncthreads();
    // A: 128 rows x 64B, async DMA (LDS stride 80B, 16B segments)
#pragma unroll
    for (int i = 0; i < 4; ++i) {
      int idx = tid + i * 128;                  // 0..511
      int m = idx >> 2, seg = idx & 3;
      async_ld16(Ybf + (size_t)(m0 + m) * Cq + k0 + seg * 8,
                 &As[m * 40 + seg * 8]);
    }
#pragma unroll
    for (int i = 0; i < 16; ++i) {
      int idx = tid + i * 128;
      int kk = idx >> 6, n = idx & 63;
      Bs[n * 40 + kk] = f2bf(W[(size_t)(k0 + kk) * Cq + n0 + n]);
    }
    wait_async0();
    __syncthreads();
    v16bf a0 = frag_a(As, 40, wave * 32, 0);
    v16bf a1 = frag_a(As, 40, wave * 32 + 16, 0);
#pragma unroll
    for (int nt = 0; nt < 4; ++nt) {
      v16bf b = frag_b(Bs, 40, nt * 16, 0);
      acc[0][nt] = wmma_bf16(a0, b, acc[0][nt]);
      acc[1][nt] = wmma_bf16(a1, b, acc[1][nt]);
    }
  }
#pragma unroll
  for (int rt = 0; rt < 2; ++rt) {
#pragma unroll
    for (int nt = 0; nt < 4; ++nt) {
      int gn = n0 + nt * 16 + (lane & 15);
#pragma unroll
      for (int e = 0; e < 8; ++e) {
        int gm = m0 + wave * 32 + rt * 16 + e + 8 * half;
        Out[(size_t)gm * Cq + gn] = acc[rt][nt][e];
      }
    }
  }
}

// ---------------------------------------------------------------------------
extern "C" void kernel_launch(void* const* d_in, const int* in_sizes, int n_in,
                              void* d_out, int out_size, void* d_ws,
                              size_t ws_size, hipStream_t stream) {
  const float* x      = (const float*)d_in[0];
  // d_in[1] = mask (causality is hardcoded in attn_kernel)
  const float* w_qkv  = (const float*)d_in[2];
  const float* w_proj = (const float*)d_in[3];
  float* out = (float*)d_out;

  const size_t elems = (size_t)Bq * Tq * Cq;  // 8,388,608
  unsigned short* Qw = (unsigned short*)d_ws; // needs 4*elems*2 = 64 MB of ws
  unsigned short* Kw = Qw + elems;
  unsigned short* Vw = Kw + elems;
  unsigned short* Yw = Vw + elems;

  dim3 blk(128);
  qkv_gemm_kernel<<<dim3(3 * Cq / 64, Bq * Tq / 128), blk, 0, stream>>>(
      x, w_qkv, Qw, Kw, Vw);
  attn_kernel<<<dim3(Tq / 64, NHq, Bq), blk, 0, stream>>>(Qw, Kw, Vw, Yw);
  proj_gemm_kernel<<<dim3(Cq / 64, Bq * Tq / 128), blk, 0, stream>>>(
      Yw, w_proj, out);
}
